// Attention_30485677867621
// MI455X (gfx1250) — compile-verified, matrix operand order checked by reading, not run on hardware
//
#include <hip/hip_runtime.h>
#include <hip/hip_bf16.h>
#include <hip/hip_fp16.h>

// ---------------------------------------------------------------------------
// Shapes (fixed by the reference)
// ---------------------------------------------------------------------------
#define BATCH   2
#define SEQ     2048
#define DIM     1024
#define HEADS   16
#define DHEAD   64
#define NROWS   (BATCH * SEQ)        // 4096
#define LN_EPS  1e-5f

typedef _Float16 v16h __attribute__((ext_vector_type(16)));
typedef _Float16 v8h  __attribute__((ext_vector_type(8)));
typedef float    v8f  __attribute__((ext_vector_type(8)));

// Concatenate two 8-half chunks into a 16-half WMMA fragment register set.
__device__ __forceinline__ v16h cat16(v8h lo, v8h hi) {
    return __builtin_shufflevector(lo, hi, 0,1,2,3,4,5,6,7,8,9,10,11,12,13,14,15);
}

// Load one 16x32 f16 A/B fragment slice for this lane.
// Per CDNA5 ISA: elements 0..7 = K{kb..kb+7}, elements 8..15 = K{kb+16..kb+23},
// where kb = 0 (lanes 0-15) or 8 (lanes 16-31). Caller passes p = rowbase + kb.
__device__ __forceinline__ v16h frag_ld(const _Float16* p) {
    v8h lo = *(const v8h*)p;
    v8h hi = *(const v8h*)(p + 16);
    return cat16(lo, hi);
}

#define WMMA_F16(a, b, c) \
    __builtin_amdgcn_wmma_f32_16x16x32_f16(false, (a), false, (b), (short)0, (c), false, false)

// ---------------------------------------------------------------------------
// Async global -> LDS tile staging (CDNA5 GLOBAL_LOAD_ASYNC_TO_LDS_B128).
// GVS addressing: mem = SADDR64 + VADDR32 + inst_offset; LDS dest byte address
// comes from the VDST VGPR (generic shared pointer low 32 bits, per ISA flat-
// aperture rule). Tracked with ASYNCcnt.
// ---------------------------------------------------------------------------
__device__ __forceinline__ void async_copy_b128(unsigned lds_byte_addr,
                                                const void* gbase,
                                                unsigned gbyte_off) {
    asm volatile("global_load_async_to_lds_b128 %0, %1, %2"
                 :: "v"(lds_byte_addr), "v"(gbyte_off), "s"(gbase)
                 : "memory");
}

__device__ __forceinline__ void wait_async0() {
    asm volatile("s_wait_asynccnt 0x0" ::: "memory");
}

// ---------------------------------------------------------------------------
// Conversion / permutation kernels
// ---------------------------------------------------------------------------
__global__ void cvt_f32_to_f16(const float* __restrict__ in,
                               _Float16* __restrict__ out, int n) {
    int i = blockIdx.x * blockDim.x + threadIdx.x;
    int stride = gridDim.x * blockDim.x;
    for (; i < n; i += stride) out[i] = (_Float16)in[i];
}

// W' rows in head-major order: out[o'][k] = W[f(o')][k], f(o') = (o'&63)*16 + (o'>>6)
__global__ void cvt_w_headperm(const float* __restrict__ in,
                               _Float16* __restrict__ out) {
    int i = blockIdx.x * blockDim.x + threadIdx.x;
    int stride = gridDim.x * blockDim.x;
    for (; i < DIM * DIM; i += stride) {
        int o = i >> 10, k = i & 1023;
        int f = ((o & 63) << 4) | (o >> 6);
        out[i] = (_Float16)in[f * DIM + k];
    }
}

// Wo' columns permuted: out[o][f'] = Wo[o][f(f')]
__global__ void cvt_wo_colperm(const float* __restrict__ in,
                               _Float16* __restrict__ out) {
    int i = blockIdx.x * blockDim.x + threadIdx.x;
    int stride = gridDim.x * blockDim.x;
    for (; i < DIM * DIM; i += stride) {
        int o = i >> 10, fp = i & 1023;
        int f = ((fp & 63) << 4) | (fp >> 6);
        out[i] = (_Float16)in[(o << 10) + f];
    }
}

// ---------------------------------------------------------------------------
// Shared NT-GEMM main loop: C(128x128) = A(128xK) * B(128xK)^T, K = 1024.
// A: (NROWS x DIM) f16 row-major.  B: (DIM x DIM) f16 row-major (N x K).
// Block = 256 threads = 8 waves laid out 2(M) x 4(N); each wave owns 64x32.
// Tiles staged via async global->LDS copies (ASYNCcnt), no VGPR round-trip.
// ---------------------------------------------------------------------------
#define LDSS 72   // LDS row stride in halfs (144B = 16B-aligned, conflict-skewed)

__device__ __forceinline__ void gemm_mainloop(const _Float16* __restrict__ A,
                                              const _Float16* __restrict__ B,
                                              int m0, int n0, v8f acc[4][2]) {
    __shared__ __align__(16) _Float16 As[128 * LDSS];
    __shared__ __align__(16) _Float16 Bs[128 * LDSS];

    const int tid  = threadIdx.x;
    const int wid  = tid >> 5;
    const int lane = tid & 31;
    const int wm   = wid & 1;       // 0..1  -> 64-row slab
    const int wn   = wid >> 1;      // 0..3  -> 32-col slab
    const int lrow = lane & 15;
    const int kb   = (lane < 16) ? 0 : 8;
    const int ldr  = tid >> 1;            // 0..127 tile row
    const int ldc  = (tid & 1) * 32;      // 0 / 32 half-row

    const unsigned lA = (unsigned)(size_t)&As[ldr * LDSS + ldc];
    const unsigned lB = (unsigned)(size_t)&Bs[ldr * LDSS + ldc];

    for (int k0 = 0; k0 < DIM; k0 += 64) {
        const unsigned gA = (unsigned)(((m0 + ldr) * DIM + k0 + ldc) * 2);
        const unsigned gB = (unsigned)(((n0 + ldr) * DIM + k0 + ldc) * 2);
        #pragma unroll
        for (int i = 0; i < 4; ++i) {
            async_copy_b128(lA + i * 16, A, gA + i * 16);
            async_copy_b128(lB + i * 16, B, gB + i * 16);
        }
        wait_async0();
        __syncthreads();
        #pragma unroll
        for (int kc = 0; kc < 64; kc += 32) {
            v16h a[4], bf[2];
            #pragma unroll
            for (int mt = 0; mt < 4; ++mt)
                a[mt] = frag_ld(&As[(wm * 64 + mt * 16 + lrow) * LDSS + kc + kb]);
            #pragma unroll
            for (int nt = 0; nt < 2; ++nt)
                bf[nt] = frag_ld(&Bs[(wn * 32 + nt * 16 + lrow) * LDSS + kc + kb]);
            #pragma unroll
            for (int mt = 0; mt < 4; ++mt)
                #pragma unroll
                for (int nt = 0; nt < 2; ++nt)
                    acc[mt][nt] = WMMA_F16(a[mt], bf[nt], acc[mt][nt]);
        }
        __syncthreads();
    }
}

// QKV projection: writes f16 in (B, H, T, Dh) layout; W rows already head-major.
__global__ __launch_bounds__(256) void gemm_qkv_kernel(
    const _Float16* __restrict__ Xh, const _Float16* __restrict__ Wh,
    const float* __restrict__ bias, _Float16* __restrict__ out) {
    const int m0 = blockIdx.y * 128, n0 = blockIdx.x * 128;
    v8f acc[4][2] = {};
    gemm_mainloop(Xh, Wh, m0, n0, acc);

    const int tid = threadIdx.x, wid = tid >> 5, lane = tid & 31;
    const int wm = wid & 1, wn = wid >> 1;
    const int rbase = (lane < 16) ? 0 : 8;
    const int cl = lane & 15;
    #pragma unroll
    for (int mt = 0; mt < 4; ++mt)
        #pragma unroll
        for (int nt = 0; nt < 2; ++nt)
            #pragma unroll
            for (int r = 0; r < 8; ++r) {
                int row = m0 + wm * 64 + mt * 16 + rbase + r;   // n = b*T + t
                int col = n0 + wn * 32 + nt * 16 + cl;          // o' = h*64 + d
                int h = col >> 6, d = col & 63;
                float v = acc[mt][nt][r] + bias[(d << 4) | h];  // bias index-permuted
                int bb = row >> 11, t = row & 2047;
                out[(((size_t)(bb * HEADS + h) * SEQ) + t) * DHEAD + d] = (_Float16)v;
            }
}

// O projection + bias + residual -> f32 scratch
__global__ __launch_bounds__(256) void gemm_oproj_kernel(
    const _Float16* __restrict__ Oh, const _Float16* __restrict__ Woh,
    const float* __restrict__ bo, const float* __restrict__ x,
    float* __restrict__ R) {
    const int m0 = blockIdx.y * 128, n0 = blockIdx.x * 128;
    v8f acc[4][2] = {};
    gemm_mainloop(Oh, Woh, m0, n0, acc);

    const int tid = threadIdx.x, wid = tid >> 5, lane = tid & 31;
    const int wm = wid & 1, wn = wid >> 1;
    const int rbase = (lane < 16) ? 0 : 8;
    const int cl = lane & 15;
    #pragma unroll
    for (int mt = 0; mt < 4; ++mt)
        #pragma unroll
        for (int nt = 0; nt < 2; ++nt)
            #pragma unroll
            for (int r = 0; r < 8; ++r) {
                int row = m0 + wm * 64 + mt * 16 + rbase + r;
                int col = n0 + wn * 32 + nt * 16 + cl;
                size_t idx = (size_t)row * DIM + col;
                R[idx] = acc[mt][nt][r] + bo[col] + x[idx];
            }
}

// ---------------------------------------------------------------------------
// Flash attention: block = 4 waves, 64 query rows for one (b, h).
// Q/K/V are f16 (B,H,T,Dh); output f16 head-major features (n, h*64+d).
// K tile staged with async copies; V transposed through VGPRs into LDS.
// ---------------------------------------------------------------------------
__global__ __launch_bounds__(128) void attn_kernel(
    const _Float16* __restrict__ Qh, const _Float16* __restrict__ Kh,
    const _Float16* __restrict__ Vh, _Float16* __restrict__ Obuf) {
    const int qblk = blockIdx.x;          // T/64 = 32
    const int bh   = blockIdx.y;          // B*H  = 32
    const int b = bh >> 4, h = bh & 15;
    const size_t base = (size_t)bh * SEQ * DHEAD;
    const _Float16* Qb = Qh + base;
    const _Float16* Kb = Kh + base;
    const _Float16* Vb = Vh + base;

    __shared__ __align__(16) _Float16 Ks[64 * LDSS];
    __shared__ __align__(16) _Float16 Vt[64 * LDSS];     // transposed: Vt[d][j]
    __shared__ __align__(16) _Float16 Ps[4][16 * LDSS];  // per-wave P staging

    const int tid  = threadIdx.x;
    const int wid  = tid >> 5;
    const int lane = tid & 31;
    const int lrow = lane & 15;
    const int kb   = (lane < 16) ? 0 : 8;
    const int q0   = qblk * 64 + wid * 16;

    // Q fragments stay in registers for the whole j-loop.
    const _Float16* Qrow = Qb + (size_t)(q0 + lrow) * DHEAD;
    v16h aq[2];
    aq[0] = frag_ld(Qrow + kb);
    aq[1] = frag_ld(Qrow + 32 + kb);

    v8f o_acc[4] = {};
    float m_r[8], l_r[8];
    #pragma unroll
    for (int r = 0; r < 8; ++r) { m_r[r] = -1e30f; l_r[r] = 0.0f; }
    const float sm_scale = 0.125f;   // 1/sqrt(64)

    const int ldr2 = tid >> 1;            // 0..63
    const int ldc2 = (tid & 1) * 32;
    const unsigned lK = (unsigned)(size_t)&Ks[ldr2 * LDSS + ldc2];

    for (int j0 = 0; j0 < SEQ; j0 += 64) {
        // Cooperative stage: K via async copy, V transposed via VGPRs.
        const unsigned gK = (unsigned)(((j0 + ldr2) * DHEAD + ldc2) * 2);
        #pragma unroll
        for (int i = 0; i < 4; ++i)
            async_copy_b128(lK + i * 16, Kb, gK + i * 16);
        #pragma unroll
        for (int i = 0; i < 4; ++i) {
            v8h vv = *(const v8h*)(Vb + (size_t)(j0 + ldr2) * DHEAD + ldc2 + i * 8);
            #pragma unroll
            for (int e = 0; e < 8; ++e)
                Vt[(ldc2 + i * 8 + e) * LDSS + ldr2] = vv[e];
        }
        wait_async0();
        __syncthreads();

        // Scores: S(16x64) = Q(16x64) x K_j^T, as 4 16x16 tiles.
        v8f s[4];
        #pragma unroll
        for (int jt = 0; jt < 4; ++jt) {
            v8f c = {};
            #pragma unroll
            for (int kc = 0; kc < 2; ++kc) {
                v16h bk = frag_ld(&Ks[(jt * 16 + lrow) * LDSS + kc * 32 + kb]);
                c = WMMA_F16(aq[kc], bk, c);
            }
            #pragma unroll
            for (int r = 0; r < 8; ++r) c[r] *= sm_scale;
            s[jt] = c;
        }

        // Online softmax. C-layout: vgpr r holds row (r | r+8), col = lane%16,
        // so per-row stats are uniform within each 16-lane half after xor-reduce.
        #pragma unroll
        for (int r = 0; r < 8; ++r) {
            float v = fmaxf(fmaxf(s[0][r], s[1][r]), fmaxf(s[2][r], s[3][r]));
            #pragma unroll
            for (int mask = 1; mask < 16; mask <<= 1)
                v = fmaxf(v, __shfl_xor(v, mask, 32));
            float mnew = fmaxf(m_r[r], v);
            float sc = __expf(m_r[r] - mnew);
            m_r[r] = mnew;
            l_r[r] *= sc;
            #pragma unroll
            for (int nt = 0; nt < 4; ++nt) o_acc[nt][r] *= sc;
        }
        #pragma unroll
        for (int jt = 0; jt < 4; ++jt)
            #pragma unroll
            for (int r = 0; r < 8; ++r)
                s[jt][r] = __expf(s[jt][r] - m_r[r]);
        #pragma unroll
        for (int r = 0; r < 8; ++r) {
            float ps = s[0][r] + s[1][r] + s[2][r] + s[3][r];
            #pragma unroll
            for (int mask = 1; mask < 16; mask <<= 1)
                ps += __shfl_xor(ps, mask, 32);
            l_r[r] += ps;
        }

        // Re-layout P: C-frag -> LDS row-major -> A-frag (per-wave, no barrier).
        _Float16* P = &Ps[wid][0];
        const int rofs = (lane < 16) ? 0 : 8;
        #pragma unroll
        for (int jt = 0; jt < 4; ++jt)
            #pragma unroll
            for (int r = 0; r < 8; ++r)
                P[(rofs + r) * LDSS + jt * 16 + lrow] = (_Float16)s[jt][r];

        v16h ap[2];
        ap[0] = frag_ld(&P[lrow * LDSS + kb]);
        ap[1] = frag_ld(&P[lrow * LDSS + 32 + kb]);

        // O += P(16x64) x V(64x64): B-frags from transposed Vt (k = j contiguous).
        #pragma unroll
        for (int nt = 0; nt < 4; ++nt)
            #pragma unroll
            for (int kc = 0; kc < 2; ++kc) {
                v16h bv = frag_ld(&Vt[(nt * 16 + lrow) * LDSS + kc * 32 + kb]);
                o_acc[nt] = WMMA_F16(ap[kc], bv, o_acc[nt]);
            }
        __syncthreads();
    }

    // Normalize and scatter to head-major feature layout (n, h*64+d).
    const int rofs = (lane < 16) ? 0 : 8;
    #pragma unroll
    for (int r = 0; r < 8; ++r) {
        float inv = 1.0f / l_r[r];
        int trow = q0 + rofs + r;
        size_t outrow = ((size_t)(b * SEQ + trow)) * DIM + h * DHEAD;
        #pragma unroll
        for (int nt = 0; nt < 4; ++nt) {
            int d = nt * 16 + lrow;
            Obuf[outrow + d] = (_Float16)(o_acc[nt][r] * inv);
        }
    }
}

// ---------------------------------------------------------------------------
// Row LayerNorm: one block per row of 1024.
// ---------------------------------------------------------------------------
__global__ __launch_bounds__(256) void ln_kernel(
    const float* __restrict__ R, const float* __restrict__ gamma,
    const float* __restrict__ beta, float* __restrict__ out) {
    const int n = blockIdx.x;
    const float* row = R + (size_t)n * DIM;
    const int tid = threadIdx.x, wid = tid >> 5, lane = tid & 31;

    float s = 0.0f, s2 = 0.0f;
    for (int i = tid; i < DIM; i += 256) {
        float v = row[i];
        s += v; s2 += v * v;
    }
    #pragma unroll
    for (int mask = 16; mask >= 1; mask >>= 1) {
        s  += __shfl_xor(s,  mask, 32);
        s2 += __shfl_xor(s2, mask, 32);
    }
    __shared__ float red[2][8];
    __shared__ float mu_s, rstd_s;
    if (lane == 0) { red[0][wid] = s; red[1][wid] = s2; }
    __syncthreads();
    if (tid == 0) {
        float S = 0.0f, S2 = 0.0f;
        #pragma unroll
        for (int w = 0; w < 8; ++w) { S += red[0][w]; S2 += red[1][w]; }
        float mu = S * (1.0f / DIM);
        float var = S2 * (1.0f / DIM) - mu * mu;
        mu_s = mu;
        rstd_s = rsqrtf(var + LN_EPS);
    }
    __syncthreads();
    float mu = mu_s, rstd = rstd_s;
    for (int i = tid; i < DIM; i += 256)
        out[(size_t)n * DIM + i] = (row[i] - mu) * rstd * gamma[i] + beta[i];
}

// ---------------------------------------------------------------------------
// Orchestration
// ---------------------------------------------------------------------------
extern "C" void kernel_launch(void* const* d_in, const int* in_sizes, int n_in,
                              void* d_out, int out_size, void* d_ws, size_t ws_size,
                              hipStream_t stream) {
    const float* x     = (const float*)d_in[0];
    const float* Wq    = (const float*)d_in[1];
    const float* bq    = (const float*)d_in[2];
    const float* Wk    = (const float*)d_in[3];
    const float* bk    = (const float*)d_in[4];
    const float* Wv    = (const float*)d_in[5];
    const float* bv    = (const float*)d_in[6];
    const float* Wo    = (const float*)d_in[7];
    const float* bo    = (const float*)d_in[8];
    const float* gamma = (const float*)d_in[9];
    const float* beta  = (const float*)d_in[10];
    float* out = (float*)d_out;

    char* ws = (char*)d_ws;
    const size_t SZ_XH = (size_t)NROWS * DIM * 2;       // 8 MB
    const size_t SZ_W  = (size_t)DIM * DIM * 2;         // 2 MB
    const size_t SZ_Q  = (size_t)BATCH * HEADS * SEQ * DHEAD * 2; // 8 MB

    _Float16* Xh   = (_Float16*)(ws);
    _Float16* Wqh  = (_Float16*)(ws + SZ_XH);
    _Float16* Wkh  = (_Float16*)(ws + SZ_XH + SZ_W);
    _Float16* Wvh  = (_Float16*)(ws + SZ_XH + 2 * SZ_W);
    _Float16* Woh  = (_Float16*)(ws + SZ_XH + 3 * SZ_W);
    _Float16* Qbuf = (_Float16*)(ws + SZ_XH + 4 * SZ_W);
    _Float16* Kbuf = (_Float16*)(ws + SZ_XH + 4 * SZ_W + SZ_Q);
    _Float16* Vbuf = (_Float16*)(ws + SZ_XH + 4 * SZ_W + 2 * SZ_Q);
    _Float16* Obuf = (_Float16*)(ws + SZ_XH + 4 * SZ_W + 3 * SZ_Q);
    float*    Rbuf = (float*)   (ws + SZ_XH + 4 * SZ_W + 4 * SZ_Q);

    // Phase 1: convert + permute
    cvt_f32_to_f16<<<2048, 256, 0, stream>>>(x, Xh, NROWS * DIM);
    cvt_w_headperm<<<1024, 256, 0, stream>>>(Wq, Wqh);
    cvt_w_headperm<<<1024, 256, 0, stream>>>(Wk, Wkh);
    cvt_w_headperm<<<1024, 256, 0, stream>>>(Wv, Wvh);
    cvt_wo_colperm<<<1024, 256, 0, stream>>>(Wo, Woh);

    // Phase 2: QKV projections (WMMA GEMM, async-staged tiles)
    dim3 gg(DIM / 128, NROWS / 128);   // (8, 32)
    gemm_qkv_kernel<<<gg, 256, 0, stream>>>(Xh, Wqh, bq, Qbuf);
    gemm_qkv_kernel<<<gg, 256, 0, stream>>>(Xh, Wkh, bk, Kbuf);
    gemm_qkv_kernel<<<gg, 256, 0, stream>>>(Xh, Wvh, bv, Vbuf);

    // Phase 3: flash attention
    attn_kernel<<<dim3(SEQ / 64, BATCH * HEADS), 128, 0, stream>>>(Qbuf, Kbuf, Vbuf, Obuf);

    // Phase 4: O projection + residual
    gemm_oproj_kernel<<<gg, 256, 0, stream>>>(Obuf, Woh, bo, x, Rbuf);

    // Phase 5: LayerNorm
    ln_kernel<<<NROWS, 256, 0, stream>>>(Rbuf, gamma, beta, out);
}
